// Net_52725018526367
// MI455X (gfx1250) — matrix operand order repaired and checked by reading, not compiled, not run on hardware
//
#include <hip/hip_runtime.h>

typedef __attribute__((ext_vector_type(16))) _Float16 v16h;
typedef __attribute__((ext_vector_type(8)))  float    v8f;
typedef __attribute__((ext_vector_type(4)))  unsigned int u32x4;
typedef _Float16 half_t;

#define N_NODES 10000
#define N_EDGES 320000
#define FDIM    512
#define HDIM    256

// ---------------------------------------------------------------- helpers
__global__ void k_f32_to_f16(half_t* __restrict__ dst, const float* __restrict__ src, long n) {
  long i = (long)blockIdx.x * blockDim.x + threadIdx.x;
  if (i < n) dst[i] = (half_t)src[i];
}

__global__ void k_zero_f32(float* __restrict__ p, long n) {
  long i = (long)blockIdx.x * blockDim.x + threadIdx.x;
  if (i < n) p[i] = 0.0f;
}

// Wt[n*Ks + k] = W[(k0+k)*256 + n]  (transpose+convert a K-slice of a weight)
__global__ void k_transpose_w(half_t* __restrict__ Wt, const float* __restrict__ W,
                              int k0, int Ks) {
  int t = blockIdx.x * blockDim.x + threadIdx.x;
  if (t >= Ks * HDIM) return;
  int n = t / Ks, k = t % Ks;
  Wt[t] = (half_t)W[(size_t)(k0 + k) * HDIM + n];
}

// ---------------------------------------------------------------- WMMA GEMM
// Out[M x 256] = A[M x K] @ W (W given transposed: Wt[256 x K] f16 row-major)
// grid.x = M/16, block = 256 (8 waves); wave w covers cols [32w, 32w+32)
// K is compile-time: the k-loop fully unrolls into one straight-line block so
// the scheduler software-pipelines the B loads against the WMMAs itself.
// Epilogue fuses: +bias, +gP[src[r]]+gQ[dst[r]], relu, +resid
template <int K>
__launch_bounds__(256)
__global__ void k_gemm(const half_t* __restrict__ A, const half_t* __restrict__ Wt,
                       const float*  __restrict__ bias,
                       const half_t* __restrict__ gP, const half_t* __restrict__ gQ,
                       const int* __restrict__ srcI, const int* __restrict__ dstI,
                       const half_t* __restrict__ resid, int do_relu,
                       half_t* __restrict__ Out) {
  constexpr int pitch = K + 8;                    // LDS pitch pad: banks + 16B align
  __shared__ half_t sA[16 * pitch];
  const int rowbase = blockIdx.x * 16;
  const int tid     = threadIdx.x;

  // stage A tile (16 rows x K halves) into LDS via the CDNA5 async path:
  // GLOBAL_LOAD_ASYNC_TO_LDS_B128 (ASYNCcnt), bypassing the VGPR file.
  constexpr int kc8 = K / 8;
#pragma unroll
  for (int c = tid; c < 16 * kc8; c += 256) {
    int row = c / kc8, c8 = c % kc8;
    const half_t* gsrc = &A[(size_t)(rowbase + row) * K + c8 * 8];
    unsigned ldsa = (unsigned)(unsigned long long)&sA[row * pitch + c8 * 8];
    asm volatile("global_load_async_to_lds_b128 %0, %1, off"
                 :: "v"(ldsa), "v"(gsrc) : "memory");
  }
  asm volatile("s_wait_asynccnt 0" ::: "memory");
  __syncthreads();

  const int w = tid >> 5, lane = tid & 31;
  const int m = lane & 15, hi = lane >> 4;
  const int n0 = w * 32 + m, n1 = n0 + 16;
  const half_t* wb0 = &Wt[(size_t)n0 * K + hi * 16];
  const half_t* wb1 = &Wt[(size_t)n1 * K + hi * 16];

  v8f acc0 = {}; v8f acc1 = {};
#pragma unroll
  for (int kc = 0; kc < K / 32; ++kc) {
    // A frag from LDS: lane(m,hi) holds k = hi*16..hi*16+15
    union { v16h v; u32x4 u[2]; } af, bf0, bf1;
    const int abase = m * pitch + kc * 32;
    af.u[0] = *(const u32x4*)&sA[abase + hi * 16];
    af.u[1] = *(const u32x4*)&sA[abase + hi * 16 + 8];
    // B frag: lane(n,hi) holds k = hi*16..hi*16+15 (contiguous in transposed Wt)
    const u32x4* p0 = (const u32x4*)(wb0 + kc * 32);
    bf0.u[0] = p0[0]; bf0.u[1] = p0[1];
    const u32x4* p1 = (const u32x4*)(wb1 + kc * 32);
    bf1.u[0] = p1[0]; bf1.u[1] = p1[1];
    acc0 = __builtin_amdgcn_wmma_f32_16x16x32_f16(false, af.v, false, bf0.v,
                                                  (short)0, acc0, false, false);
    acc1 = __builtin_amdgcn_wmma_f32_16x16x32_f16(false, af.v, false, bf1.v,
                                                  (short)0, acc1, false, false);
  }

  const float b0 = bias ? bias[n0] : 0.0f;
  const float b1 = bias ? bias[n1] : 0.0f;
#pragma unroll
  for (int i = 0; i < 8; ++i) {                  // C layout: VGPR i -> row i+8*hi
    const int r = rowbase + i + 8 * hi;
    float v0 = acc0[i] + b0;
    float v1 = acc1[i] + b1;
    if (gP) {
      const int s = srcI[r], d = dstI[r];
      v0 += (float)gP[(size_t)s * HDIM + n0] + (float)gQ[(size_t)d * HDIM + n0];
      v1 += (float)gP[(size_t)s * HDIM + n1] + (float)gQ[(size_t)d * HDIM + n1];
    }
    if (do_relu) { v0 = v0 > 0.f ? v0 : 0.f; v1 = v1 > 0.f ? v1 : 0.f; }
    if (resid) {
      v0 += (float)resid[(size_t)r * HDIM + n0];
      v1 += (float)resid[(size_t)r * HDIM + n1];
    }
    Out[(size_t)r * HDIM + n0] = (half_t)v0;
    Out[(size_t)r * HDIM + n1] = (half_t)v1;
  }
}

// ------------------------------------------------- conv_in edge hidden layer
// h1[e][n] = relu(P[src][n] + Q[dst][n] + ea[e][0:6] @ Wtail[:,n] + be[n])
__launch_bounds__(256)
__global__ void k_edge_h1(const half_t* __restrict__ Ph, const half_t* __restrict__ Qh,
                          const float* __restrict__ ea, const float* __restrict__ Wtail,
                          const float* __restrict__ be,
                          const int* __restrict__ srcI, const int* __restrict__ dstI,
                          half_t* __restrict__ H1) {
  const int e = blockIdx.x, n = threadIdx.x;
  const int s = srcI[e], d = dstI[e];
  float acc = be[n] + (float)Ph[(size_t)s * HDIM + n] + (float)Qh[(size_t)d * HDIM + n];
#pragma unroll
  for (int j = 0; j < 6; ++j)
    acc += ea[(size_t)e * 6 + j] * Wtail[j * HDIM + n];
  H1[(size_t)e * HDIM + n] = (half_t)(acc > 0.f ? acc : 0.f);
}

// ---------------------------------------------------------- segment mean agg
__global__ void k_agg_atomic(const half_t* __restrict__ EA1, const int* __restrict__ dstI,
                             float* __restrict__ AGG) {
  size_t t = (size_t)blockIdx.x * 256 + threadIdx.x;  // t < E*256 exactly
  int e = (int)(t >> 8), h = (int)(t & 255);
  atomicAdd(&AGG[(size_t)dstI[e] * HDIM + h], (float)EA1[t]);
}
__global__ void k_deg(const int* __restrict__ dstI, float* __restrict__ deg) {
  int e = blockIdx.x * 256 + threadIdx.x;
  if (e < N_EDGES) atomicAdd(&deg[dstI[e]], 1.0f);
}
__global__ void k_agg_scale(const float* __restrict__ AGG, const float* __restrict__ deg,
                            half_t* __restrict__ AGGh) {
  size_t t = (size_t)blockIdx.x * 256 + threadIdx.x;  // t < N*256 exactly
  AGGh[t] = (half_t)(AGG[t] / fmaxf(deg[t >> 8], 1.0f));
}
__global__ void k_concat(const half_t* __restrict__ Xh, const half_t* __restrict__ AGGh,
                         half_t* __restrict__ XCAT) {
  size_t t = (size_t)blockIdx.x * 256 + threadIdx.x;  // t < N*768 exactly
  int r = (int)(t / 768), c = (int)(t % 768);
  XCAT[t] = (c < FDIM) ? Xh[(size_t)r * FDIM + c] : AGGh[(size_t)r * HDIM + (c - FDIM)];
}

// ------------------------------------------------------- fused edge predictor
// out[e] = leaky_relu(EA2[e]@Wp1 + bp1) @ Wp2 + bp2   (h never hits memory)
// grid = E/128, block 256: wave w handles 16-row tile w, sweeps all 16 n-tiles.
// Fully unrolled: 128 straight-line WMMAs, A frags pinned in 64 VGPRs.
__launch_bounds__(256)
__global__ void k_predictor(const half_t* __restrict__ EA2, const half_t* __restrict__ Wt,
                            const float* __restrict__ bp1, const float* __restrict__ Wp2,
                            const float* __restrict__ bp2, float* __restrict__ Out) {
  const int w = threadIdx.x >> 5, lane = threadIdx.x & 31;
  const int m = lane & 15, hi = lane >> 4;
  const int rowbase = blockIdx.x * 128 + w * 16;

  union BF { v16h v; u32x4 u[2]; };
  BF af[8];
  const half_t* Arow = EA2 + (size_t)(rowbase + m) * HDIM;
#pragma unroll
  for (int kc = 0; kc < 8; ++kc) {
    af[kc].u[0] = *(const u32x4*)&Arow[kc * 32 + hi * 16];
    af[kc].u[1] = *(const u32x4*)&Arow[kc * 32 + hi * 16 + 8];
  }

  float rowsum[8] = {0, 0, 0, 0, 0, 0, 0, 0};
#pragma unroll
  for (int nt = 0; nt < 16; ++nt) {
    v8f acc = {};
    const int n = nt * 16 + m;
    const half_t* Wrow = Wt + (size_t)n * HDIM;
#pragma unroll
    for (int kc = 0; kc < 8; ++kc) {
      BF bf;
      const u32x4* p = (const u32x4*)&Wrow[kc * 32 + hi * 16];
      bf.u[0] = p[0]; bf.u[1] = p[1];
      acc = __builtin_amdgcn_wmma_f32_16x16x32_f16(false, af[kc].v, false, bf.v,
                                                   (short)0, acc, false, false);
    }
    const float b = bp1[n], w2 = Wp2[n];
#pragma unroll
    for (int i = 0; i < 8; ++i) {
      float h = acc[i] + b;
      h = h > 0.f ? h : 0.01f * h;             // leaky_relu, slope 0.01
      rowsum[i] += h * w2;
    }
  }
#pragma unroll
  for (int i = 0; i < 8; ++i) {                // reduce over n within 16-lane halves
    float v = rowsum[i];
    v += __shfl_xor(v, 1, 32);
    v += __shfl_xor(v, 2, 32);
    v += __shfl_xor(v, 4, 32);
    v += __shfl_xor(v, 8, 32);
    if (m == 0) Out[rowbase + i + 8 * hi] = v + bp2[0];
  }
}

// ---------------------------------------------------------------- launcher
extern "C" void kernel_launch(void* const* d_in, const int* in_sizes, int n_in,
                              void* d_out, int out_size, void* d_ws, size_t ws_size,
                              hipStream_t stream) {
  (void)in_sizes; (void)n_in; (void)out_size; (void)ws_size;
  const float* x    = (const float*)d_in[0];
  const int*   eix  = (const int*)  d_in[1];
  const float* ea   = (const float*)d_in[2];
  const float* We0a = (const float*)d_in[3];  const float* be0a = (const float*)d_in[4];
  const float* We0b = (const float*)d_in[5];  const float* be0b = (const float*)d_in[6];
  const float* Wn0a = (const float*)d_in[7];  const float* bn0a = (const float*)d_in[8];
  const float* Wn0b = (const float*)d_in[9];  const float* bn0b = (const float*)d_in[10];
  const float* We1a = (const float*)d_in[11]; const float* be1a = (const float*)d_in[12];
  const float* We1b = (const float*)d_in[13]; const float* be1b = (const float*)d_in[14];
  // Wn1a/Wn1b (d_in[15..18]) are dead: x2 never feeds the output.
  const float* Wp1  = (const float*)d_in[19]; const float* bp1  = (const float*)d_in[20];
  const float* Wp2  = (const float*)d_in[21]; const float* bp2  = (const float*)d_in[22];
  const int* srcI = eix;
  const int* dstI = eix + N_EDGES;
  float* out = (float*)d_out;

  char* ws = (char*)d_ws; size_t off = 0;
  auto alloc = [&](size_t bytes) -> char* {
    char* p = ws + off; off += (bytes + 255) & ~(size_t)255; return p;
  };
  half_t* bufE1 = (half_t*)alloc((size_t)N_EDGES * HDIM * 2);  // H1 -> EA2
  half_t* bufE2 = (half_t*)alloc((size_t)N_EDGES * HDIM * 2);  // EA1
  half_t* bufE3 = (half_t*)alloc((size_t)N_EDGES * HDIM * 2);  // HC
  half_t* Xh    = (half_t*)alloc((size_t)N_NODES * FDIM * 2);
  half_t* XCAT  = (half_t*)alloc((size_t)N_NODES * 768 * 2);
  half_t* Ph    = (half_t*)alloc((size_t)N_NODES * HDIM * 2);
  half_t* Qh    = (half_t*)alloc((size_t)N_NODES * HDIM * 2);
  half_t* NHh   = (half_t*)alloc((size_t)N_NODES * HDIM * 2);
  half_t* X1h   = (half_t*)alloc((size_t)N_NODES * HDIM * 2);
  half_t* P1h   = (half_t*)alloc((size_t)N_NODES * HDIM * 2);
  half_t* Q1h   = (half_t*)alloc((size_t)N_NODES * HDIM * 2);
  float*  AGGf  = (float*) alloc((size_t)N_NODES * HDIM * 4);
  half_t* AGGh  = (half_t*)alloc((size_t)N_NODES * HDIM * 2);
  float*  degf  = (float*) alloc((size_t)N_NODES * 4);
  half_t* WtP   = (half_t*)alloc((size_t)HDIM * 512 * 2);
  half_t* WtQ   = (half_t*)alloc((size_t)HDIM * 512 * 2);
  half_t* WtN0a = (half_t*)alloc((size_t)HDIM * 768 * 2);
  half_t* WtE0b = (half_t*)alloc((size_t)HDIM * 256 * 2);
  half_t* WtN0b = (half_t*)alloc((size_t)HDIM * 256 * 2);
  half_t* WtP1  = (half_t*)alloc((size_t)HDIM * 256 * 2);
  half_t* WtQ1  = (half_t*)alloc((size_t)HDIM * 256 * 2);
  half_t* WtG   = (half_t*)alloc((size_t)HDIM * 256 * 2);
  half_t* WtE1b = (half_t*)alloc((size_t)HDIM * 256 * 2);
  half_t* WtWp1 = (half_t*)alloc((size_t)HDIM * 256 * 2);

  const half_t* NH = nullptr; const int* NI = nullptr;  // null epilogue args

  // ---- weight prep + input convert
  {
    long n = (long)N_NODES * FDIM;
    k_f32_to_f16<<<(unsigned)((n + 255) / 256), 256, 0, stream>>>(Xh, x, n);
  }
  k_transpose_w<<<512, 256, 0, stream>>>(WtP,   We0a,   0, 512);
  k_transpose_w<<<512, 256, 0, stream>>>(WtQ,   We0a, 512, 512);
  k_transpose_w<<<768, 256, 0, stream>>>(WtN0a, Wn0a,   0, 768);
  k_transpose_w<<<256, 256, 0, stream>>>(WtE0b, We0b,   0, 256);
  k_transpose_w<<<256, 256, 0, stream>>>(WtN0b, Wn0b,   0, 256);
  k_transpose_w<<<256, 256, 0, stream>>>(WtP1,  We1a,   0, 256);
  k_transpose_w<<<256, 256, 0, stream>>>(WtQ1,  We1a, 256, 256);
  k_transpose_w<<<256, 256, 0, stream>>>(WtG,   We1a, 512, 256);
  k_transpose_w<<<256, 256, 0, stream>>>(WtE1b, We1b,   0, 256);
  k_transpose_w<<<256, 256, 0, stream>>>(WtWp1, Wp1,    0, 256);

  // ---- conv_in: node-level precompute P = x@We0a[0:512], Q = x@We0a[512:1024]
  k_gemm<512><<<N_NODES / 16, 256, 0, stream>>>(Xh, WtP, nullptr, NH, NH, NI, NI, NH, 0, Ph);
  k_gemm<512><<<N_NODES / 16, 256, 0, stream>>>(Xh, WtQ, nullptr, NH, NH, NI, NI, NH, 0, Qh);

  // ---- conv_in edge MLP
  k_edge_h1<<<N_EDGES, 256, 0, stream>>>(Ph, Qh, ea, We0a + 1024 * HDIM, be0a,
                                         srcI, dstI, bufE1);                       // H1
  k_gemm<256><<<N_EDGES / 16, 256, 0, stream>>>(bufE1, WtE0b, be0b,
                                                NH, NH, NI, NI, NH, 0, bufE2);     // EA1

  // ---- conv_in node MLP (mean aggregation)
  k_zero_f32<<<(N_NODES * HDIM) / 256, 256, 0, stream>>>(AGGf, (long)N_NODES * HDIM);
  k_zero_f32<<<(N_NODES + 255) / 256, 256, 0, stream>>>(degf, N_NODES);
  k_agg_atomic<<<N_EDGES, 256, 0, stream>>>(bufE2, dstI, AGGf);
  k_deg<<<(N_EDGES + 255) / 256, 256, 0, stream>>>(dstI, degf);
  k_agg_scale<<<(N_NODES * HDIM) / 256, 256, 0, stream>>>(AGGf, degf, AGGh);
  k_concat<<<(N_NODES * 768) / 256, 256, 0, stream>>>(Xh, AGGh, XCAT);
  k_gemm<768><<<N_NODES / 16, 256, 0, stream>>>(XCAT, WtN0a, bn0a,
                                                NH, NH, NI, NI, NH, 1, NHh);
  k_gemm<256><<<N_NODES / 16, 256, 0, stream>>>(NHh, WtN0b, bn0b,
                                                NH, NH, NI, NI, NH, 0, X1h);       // x1

  // ---- conv[0] edge MLP (node model skipped: dead code)
  k_gemm<256><<<N_NODES / 16, 256, 0, stream>>>(X1h, WtP1, nullptr, NH, NH, NI, NI, NH, 0, P1h);
  k_gemm<256><<<N_NODES / 16, 256, 0, stream>>>(X1h, WtQ1, nullptr, NH, NH, NI, NI, NH, 0, Q1h);
  k_gemm<256><<<N_EDGES / 16, 256, 0, stream>>>(bufE2, WtG, be1a,
                                                P1h, Q1h, srcI, dstI, NH, 1, bufE3);  // HC
  k_gemm<256><<<N_EDGES / 16, 256, 0, stream>>>(bufE3, WtE1b, be1b,
                                                NH, NH, NI, NI, bufE2, 0, bufE1);     // EA2

  // ---- predictor (fully fused)
  k_predictor<<<N_EDGES / 128, 256, 0, stream>>>(bufE1, WtWp1, bp1, Wp2, bp2, out);
}
